// PHD_DecoderBlock_25125558682004
// MI455X (gfx1250) — matrix operand order repaired and checked by reading, not compiled, not run on hardware
//
#include <hip/hip_runtime.h>
#include <hip/hip_bf16.h>
#include <stdint.h>

// ---------------------------------------------------------------------------
// PHD_DecoderBlock for MI455X (gfx1250, wave32, WMMA).
// All projections use v_wmma_f32_16x16x32_bf16 (bf16 in, f32 accum).
// 4 Mamba directions are batched into one pipeline (M = 4*B*L = 131072 rows).
// GEMM staging uses GLOBAL_LOAD_ASYNC_TO_LDS_B128 (ASYNCcnt-tracked DMA) —
// builtin confirmed present; falls back to reg-staged LDS if unavailable.
// ---------------------------------------------------------------------------

typedef __bf16 bf16_t;
typedef bf16_t v16bf __attribute__((ext_vector_type(16)));
typedef float  v8f   __attribute__((ext_vector_type(8)));

#define EPS 1e-5f

// Problem constants (match reference)
#define Bn   8
#define CIN  256
#define Cc   128
#define Hh   64
#define Ww   64
#define HW   4096
#define DI   256
#define Nst  16
#define Rr   8
#define Kk   7
#define DC   4
#define MID  8
#define Ltok 4096
#define MTOK (Bn*HW)          // 32768
#define MSEQ (4*Bn*Ltok)      // 131072

// gfx1250 async global->LDS path (guarded: host pass / older clang fall back)
#if __has_builtin(__builtin_amdgcn_global_load_async_to_lds_b128) && \
    __has_builtin(__builtin_amdgcn_s_wait_asynccnt)
#define USE_ASYNC_LDS 1
#else
#define USE_ASYNC_LDS 0
#endif

#if USE_ASYNC_LDS
// builtin signature (from hipcc diagnostic): params are pointers to
// int __attribute__((vector_size(16))) in AS(1) (global) and AS(3) (LDS).
typedef int vi4_t __attribute__((vector_size(16)));
typedef __attribute__((address_space(1))) vi4_t* gvi4_t;
typedef __attribute__((address_space(3))) vi4_t* lvi4_t;
// copy 32 contiguous bytes global -> LDS via two async b128 transfers
__device__ __forceinline__ void async_cp32(const void* g, void* l) {
    gvi4_t gp = (gvi4_t)(uintptr_t)g;
    lvi4_t lp = (lvi4_t)(uint32_t)(uintptr_t)l;
    __builtin_amdgcn_global_load_async_to_lds_b128(gp, lp, 0, 0);
    __builtin_amdgcn_global_load_async_to_lds_b128(gp, lp, 16, 0);
}
#endif

// ----------------------------- small helpers -------------------------------

__device__ __forceinline__ int perm_dir(int dir, int t) {
    // involutive token permutations for the 4 scan directions
    int h = t >> 6, w = t & 63;
    switch (dir) {
        case 0:  return t;
        case 1:  return (HW - 1) - t;
        case 2:  return (w << 6) | h;
        default: return (HW - 1) - ((w << 6) | h);
    }
}

__device__ __forceinline__ float siluf(float x) {
    return x / (1.f + __expf(-x));
}

// ------------------------------ prep kernels -------------------------------

__global__ void k_bn_fold(const float* __restrict__ g, const float* __restrict__ b,
                          const float* __restrict__ m, const float* __restrict__ v,
                          float* __restrict__ sc, float* __restrict__ bi, int n) {
    int i = blockIdx.x * blockDim.x + threadIdx.x;
    if (i < n) {
        float s = g[i] * rsqrtf(v[i] + EPS);
        sc[i] = s;
        bi[i] = b[i] - m[i] * s;
    }
}

// straight f32 -> bf16 convert (weight already stored [N][K])
__global__ void k_cvt(const float* __restrict__ src, bf16_t* __restrict__ dst, int n) {
    int i = blockIdx.x * blockDim.x + threadIdx.x;
    if (i < n) dst[i] = (bf16_t)src[i];
}

// transpose+convert: src[rows][cols] f32 -> dst[cols][rows] bf16
__global__ void k_cvt_T(const float* __restrict__ src, bf16_t* __restrict__ dst,
                        int rows, int cols) {
    int i = blockIdx.x * blockDim.x + threadIdx.x;
    if (i < rows * cols) {
        int r = i / cols, c = i % cols;
        dst[(size_t)c * rows + r] = (bf16_t)src[i];
    }
}

// NCHW f32 -> token-major bf16 rows [B*HW][CIN]
__global__ void k_nchw2rows(const float* __restrict__ x, bf16_t* __restrict__ rows) {
    int idx = blockIdx.x * blockDim.x + threadIdx.x;   // (b, hw, c), c fastest
    int c  = idx & (CIN - 1);
    int hw = (idx >> 8) & (HW - 1);
    int b  = idx >> 20;
    rows[idx] = (bf16_t)x[((size_t)(b * CIN + c)) * HW + hw];
}

// ----------------------------- WMMA GEMM -----------------------------------
// C[M][N] (+epilogue) = A[M][K] * Bt[N][K]^T, A/Bt bf16 row-major.
// Block tile 128x64, K-step 32, 8 waves each computing 32x32 (2x2 frags).

#define BM 128
#define BN 64
#define BK 32
#define LDAs (BK + 8)
#define LDBs (BK + 8)

enum { EPI_NONE = 0, EPI_BNRELU = 1, EPI_BIAS = 2 };

union FragU { uint4 u[2]; v16bf v; };

template<int EPI, bool OUT_BF16>
__global__ __launch_bounds__(256)
void gemm_bf16_k(const bf16_t* __restrict__ A, const bf16_t* __restrict__ Bt,
                 void* __restrict__ Cptr, const float* __restrict__ scale,
                 const float* __restrict__ bias, int M, int N, int K) {
    __shared__ __align__(16) bf16_t As[2][BM][LDAs];
    __shared__ __align__(16) bf16_t Bs[2][BN][LDBs];

    const int tid  = threadIdx.x;
    const int mblk = blockIdx.x * BM;
    const int nblk = blockIdx.y * BN;
    const int wave = tid >> 5, lane = tid & 31;
    const int wm = wave >> 1, wn = wave & 1;          // 4x2 wave grid
    const int lh = lane >> 4, lr = lane & 15;

    // global tile loaders: A 128x32 (256 thr x 16 halfs), B 64x32 (128 thr x 16)
    const int arow = tid >> 1, acol = (tid & 1) * 16;
    const int brow = (tid & 127) >> 1, bcol = (tid & 1) * 16;
    const bool bldr = tid < 128;

    const int nk = K / BK;

#if USE_ASYNC_LDS
    { // initial async fill of buffer 0 (ASYNCcnt-tracked DMA into LDS)
        const bf16_t* ag = A + (size_t)(mblk + arow) * K + acol;
        async_cp32(ag, &As[0][arow][acol]);
        if (bldr && (nblk + brow < N)) {
            const bf16_t* bg = Bt + (size_t)(nblk + brow) * K + bcol;
            async_cp32(bg, &Bs[0][brow][bcol]);
        }
        __builtin_amdgcn_s_wait_asynccnt(0);
    }
#else
    { // initial fill of buffer 0 (reg staged)
        const bf16_t* ag = A + (size_t)(mblk + arow) * K + acol;
        uint4 a0 = ((const uint4*)ag)[0];
        uint4 a1 = ((const uint4*)ag)[1];
        *(uint4*)&As[0][arow][acol]     = a0;
        *(uint4*)&As[0][arow][acol + 8] = a1;
        if (bldr) {
            uint4 b0 = {0,0,0,0}, b1 = {0,0,0,0};
            if (nblk + brow < N) {
                const bf16_t* bg = Bt + (size_t)(nblk + brow) * K + bcol;
                b0 = ((const uint4*)bg)[0];
                b1 = ((const uint4*)bg)[1];
            }
            *(uint4*)&Bs[0][brow][bcol]     = b0;
            *(uint4*)&Bs[0][brow][bcol + 8] = b1;
        }
    }
#endif
    __syncthreads();

    v8f acc00 = {}, acc01 = {}, acc10 = {}, acc11 = {};

    for (int kt = 0; kt < nk; ++kt) {
        const int buf = kt & 1;
        const bool more = (kt + 1 < nk);

#if USE_ASYNC_LDS
        if (more) { // DMA next K-tile into the other buffer while we compute
            const int k0 = (kt + 1) * BK;
            const bf16_t* ag = A + (size_t)(mblk + arow) * K + k0 + acol;
            async_cp32(ag, &As[buf ^ 1][arow][acol]);
            if (bldr && (nblk + brow < N)) {
                const bf16_t* bg = Bt + (size_t)(nblk + brow) * K + k0 + bcol;
                async_cp32(bg, &Bs[buf ^ 1][brow][bcol]);
            }
        }
#else
        uint4 na0, na1, nb0 = {0,0,0,0}, nb1 = {0,0,0,0};
        if (more) {
            const int k0 = (kt + 1) * BK;
            const bf16_t* ag = A + (size_t)(mblk + arow) * K + k0 + acol;
            na0 = ((const uint4*)ag)[0];
            na1 = ((const uint4*)ag)[1];
            if (bldr && (nblk + brow < N)) {
                const bf16_t* bg = Bt + (size_t)(nblk + brow) * K + k0 + bcol;
                nb0 = ((const uint4*)bg)[0];
                nb1 = ((const uint4*)bg)[1];
            }
            if (kt + 2 < nk)  // keep the next-next A tile flowing toward the WGP
                __builtin_prefetch(A + (size_t)(mblk + arow) * K + (kt + 2) * BK + acol, 0, 3);
        }
#endif

        // fragment loads per ISA 7.12.2 (16-bit A 16x32 / B 32x16, wave32)
        FragU fa0, fa1, fb0, fb1;
        fa0.u[0] = *(const uint4*)&As[buf][wm * 32 + lr][lh * 8];
        fa0.u[1] = *(const uint4*)&As[buf][wm * 32 + lr][16 + lh * 8];
        fa1.u[0] = *(const uint4*)&As[buf][wm * 32 + 16 + lr][lh * 8];
        fa1.u[1] = *(const uint4*)&As[buf][wm * 32 + 16 + lr][16 + lh * 8];
        fb0.u[0] = *(const uint4*)&Bs[buf][wn * 32 + lr][lh * 8];
        fb0.u[1] = *(const uint4*)&Bs[buf][wn * 32 + lr][16 + lh * 8];
        fb1.u[0] = *(const uint4*)&Bs[buf][wn * 32 + 16 + lr][lh * 8];
        fb1.u[1] = *(const uint4*)&Bs[buf][wn * 32 + 16 + lr][16 + lh * 8];

        acc00 = __builtin_amdgcn_wmma_f32_16x16x32_bf16(false, fa0.v, false, fb0.v,
                                                        (short)0, acc00, false, false);
        acc01 = __builtin_amdgcn_wmma_f32_16x16x32_bf16(false, fa0.v, false, fb1.v,
                                                        (short)0, acc01, false, false);
        acc10 = __builtin_amdgcn_wmma_f32_16x16x32_bf16(false, fa1.v, false, fb0.v,
                                                        (short)0, acc10, false, false);
        acc11 = __builtin_amdgcn_wmma_f32_16x16x32_bf16(false, fa1.v, false, fb1.v,
                                                        (short)0, acc11, false, false);

#if USE_ASYNC_LDS
        if (more) __builtin_amdgcn_s_wait_asynccnt(0);
#else
        if (more) {
            const int nb = buf ^ 1;
            *(uint4*)&As[nb][arow][acol]     = na0;
            *(uint4*)&As[nb][arow][acol + 8] = na1;
            if (bldr) {
                *(uint4*)&Bs[nb][brow][bcol]     = nb0;
                *(uint4*)&Bs[nb][brow][bcol + 8] = nb1;
            }
        }
#endif
        __syncthreads();
    }

    // epilogue: C/D layout — VGPR i: lanes 0-15 -> M=i, lanes 16-31 -> M=8+i
    const v8f accs[4] = { acc00, acc01, acc10, acc11 };
    #pragma unroll
    for (int f = 0; f < 4; ++f) {
        const int fi = f >> 1, fj = f & 1;
        const int row0 = mblk + wm * 32 + fi * 16 + lh * 8;
        const int col  = nblk + wn * 32 + fj * 16 + lr;
        if (col >= N) continue;
        float sc = 1.f, bi = 0.f;
        if (EPI == EPI_BNRELU) { sc = scale[col]; bi = bias[col]; }
        else if (EPI == EPI_BIAS) { bi = bias[col]; }
        #pragma unroll
        for (int i = 0; i < 8; ++i) {
            float val = accs[f][i];
            if (EPI == EPI_BNRELU) { val = val * sc + bi; val = val > 0.f ? val : 0.f; }
            else if (EPI == EPI_BIAS) { val += bi; }
            const size_t o = (size_t)(row0 + i) * N + col;
            if (OUT_BF16) ((bf16_t*)Cptr)[o] = (bf16_t)val;
            else          ((float*)Cptr)[o]  = val;
        }
    }
}

// ------------------------- depthwise 7-tap conv ----------------------------
// hv = ReLU(BN(dw1x7(p))) + ReLU(BN(dw7x1(p))), token-major bf16

__global__ void k_dwconv(const bf16_t* __restrict__ p,
                         const float* __restrict__ shw, const float* __restrict__ svw,
                         const float* __restrict__ shs, const float* __restrict__ shb,
                         const float* __restrict__ svs, const float* __restrict__ svb,
                         bf16_t* __restrict__ hv) {
    int idx = blockIdx.x * blockDim.x + threadIdx.x;
    int c  = idx & (Cc - 1);
    int hw = (idx >> 7) & (HW - 1);
    int b  = idx >> 19;
    int h = hw >> 6, w = hw & 63;
    float ah = 0.f, av = 0.f;
    #pragma unroll
    for (int j = 0; j < Kk; ++j) {
        int ww2 = w - 3 + j;
        if (ww2 >= 0 && ww2 < Ww)
            ah += shw[c * Kk + j] * (float)p[(((size_t)b * HW + (h * Ww + ww2)) << 7) + c];
        int hh2 = h - 3 + j;
        if (hh2 >= 0 && hh2 < Hh)
            av += svw[c * Kk + j] * (float)p[(((size_t)b * HW + (hh2 * Ww + w)) << 7) + c];
    }
    ah = ah * shs[c] + shb[c]; ah = ah > 0.f ? ah : 0.f;
    av = av * svs[c] + svb[c]; av = av > 0.f ? av : 0.f;
    hv[idx] = (bf16_t)(ah + av);
}

// ------------------- gather (4 directions) + LayerNorm ---------------------
// xn[(dir*B+b)*L + t][c] = LN(xr[b*L + perm_dir(t)][:])  — one wave per token

__global__ __launch_bounds__(256)
void k_gather_ln(const bf16_t* __restrict__ xr, const float* __restrict__ g,
                 const float* __restrict__ bb, bf16_t* __restrict__ xn) {
    const int wave = threadIdx.x >> 5, lane = threadIdx.x & 31;
    const int R = blockIdx.x * 8 + wave;            // [0, MSEQ)
    const int dir = R >> 15;
    const int rem = R & 32767;
    const int b = rem >> 12, t = rem & 4095;
    const size_t src = (size_t)(b * Ltok + perm_dir(dir, t)) * Cc;
    float v[4], s = 0.f, q = 0.f;
    #pragma unroll
    for (int i = 0; i < 4; ++i) {
        v[i] = (float)xr[src + lane + 32 * i];
        s += v[i]; q += v[i] * v[i];
    }
    #pragma unroll
    for (int off = 16; off > 0; off >>= 1) {
        s += __shfl_xor(s, off, 32);
        q += __shfl_xor(q, off, 32);
    }
    const float mu  = s * (1.f / Cc);
    const float var = q * (1.f / Cc) - mu * mu;
    const float inv = rsqrtf(var + EPS);
    #pragma unroll
    for (int i = 0; i < 4; ++i) {
        const int c = lane + 32 * i;
        xn[(size_t)R * Cc + c] = (bf16_t)((v[i] - mu) * inv * g[c] + bb[c]);
    }
}

// ------------------- causal depthwise conv1d + SiLU ------------------------

__global__ void k_conv1d(const bf16_t* __restrict__ xz, const float* __restrict__ cw,
                         const float* __restrict__ cb, bf16_t* __restrict__ xc) {
    int idx = blockIdx.x * blockDim.x + threadIdx.x;   // (R, d), d fastest
    int d = idx & (DI - 1);
    int R = idx >> 8;
    int t = R & (Ltok - 1);
    float a = cb[d];
    #pragma unroll
    for (int j = 0; j < DC; ++j) {
        int tj = t - 3 + j;
        if (tj >= 0) a += cw[d * DC + j] * (float)xz[((size_t)(R - 3 + j)) * (2 * DI) + d];
    }
    xc[(size_t)R * DI + d] = (bf16_t)siluf(a);
}

// ------------------------------- SSM scan ----------------------------------
// 32 blocks = (dir, b); 256 threads = channel d; 16-state in registers.
// dt computed in-register from the staged x-proj rows; silu(z) gate fused.

__global__ __launch_bounds__(256)
void k_scan(const float* __restrict__ dbl, const bf16_t* __restrict__ xc,
            const bf16_t* __restrict__ xz, const float* __restrict__ dtw_,
            const float* __restrict__ dtb_, const float* __restrict__ Alog,
            const float* __restrict__ Dp, bf16_t* __restrict__ yg) {
    __shared__ float dch[64][Rr + 2 * Nst];            // 64 steps x 40 floats
    const int d = threadIdx.x;
    const size_t Rbase = (size_t)blockIdx.x * Ltok;
    float Ad[Nst], dtw[Rr], h[Nst];
    #pragma unroll
    for (int n = 0; n < Nst; ++n) { Ad[n] = -__expf(Alog[d * Nst + n]); h[n] = 0.f; }
    #pragma unroll
    for (int r = 0; r < Rr; ++r) dtw[r] = dtw_[r * DI + d];
    const float dtb = dtb_[d], Dd = Dp[d];

    for (int ch = 0; ch < Ltok / 64; ++ch) {
        __syncthreads();
        for (int i = threadIdx.x; i < 64 * 40; i += 256)
            dch[i / 40][i % 40] = dbl[(Rbase + ch * 64 + i / 40) * 40 + i % 40];
        __syncthreads();
        for (int tt = 0; tt < 64; ++tt) {
            const size_t R = Rbase + ch * 64 + tt;
            const float u = (float)xc[R * DI + d];
            float raw = dtb;
            #pragma unroll
            for (int r = 0; r < Rr; ++r) raw += dch[tt][r] * dtw[r];
            const float dt = raw > 20.f ? raw : log1pf(__expf(raw));
            const float du = dt * u;
            float y = 0.f;
            #pragma unroll
            for (int n = 0; n < Nst; ++n) {
                h[n] = h[n] * __expf(dt * Ad[n]) + du * dch[tt][Rr + n];
                y += h[n] * dch[tt][Rr + Nst + n];
            }
            y += u * Dd;
            const float z = (float)xz[R * (2 * DI) + DI + d];
            yg[R * DI + d] = (bf16_t)(y * siluf(z));
        }
    }
}

// ------------- 4-direction scatter + average + residual --------------------

__global__ void k_feat_global(const float* __restrict__ yout,
                              const bf16_t* __restrict__ xr, float* __restrict__ fg) {
    int idx = blockIdx.x * blockDim.x + threadIdx.x;   // (b, p, c), c fastest
    int c = idx & (Cc - 1);
    int p = (idx >> 7) & (HW - 1);
    int b = idx >> 19;
    float acc = 0.f;
    #pragma unroll
    for (int dir = 0; dir < 4; ++dir)
        acc += yout[(((size_t)(dir * Bn + b)) * Ltok + perm_dir(dir, p)) * Cc + c];
    fg[idx] = acc * 0.25f + (float)xr[((size_t)(b * HW + p)) * Cc + c];
}

// ------------------------- SE head + combine -------------------------------

__global__ void k_smean(const float* __restrict__ fl, const float* __restrict__ fg,
                        float* __restrict__ s) {
    __shared__ float red[256];
    const int bc = blockIdx.x;                         // b*C + c
    const int b = bc >> 7, c = bc & (Cc - 1);
    float a = 0.f;
    for (int hw = threadIdx.x; hw < HW; hw += 256) {
        const size_t i = ((size_t)(b * HW + hw)) * Cc + c;
        a += fl[i] + fg[i];
    }
    red[threadIdx.x] = a;
    __syncthreads();
    for (int st = 128; st > 0; st >>= 1) {
        if (threadIdx.x < st) red[threadIdx.x] += red[threadIdx.x + st];
        __syncthreads();
    }
    if (threadIdx.x == 0) s[bc] = red[0] * (1.f / HW);
}

__global__ void k_se(const float* __restrict__ s, const float* __restrict__ sk1,
                     const float* __restrict__ sk2, float* __restrict__ wgt) {
    __shared__ float mid[Bn][MID];
    const int tid = threadIdx.x;
    if (tid < Bn * MID) {
        const int b = tid >> 3, j = tid & 7;
        float a = 0.f;
        for (int c = 0; c < Cc; ++c) a += s[b * Cc + c] * sk1[c * MID + j];
        mid[b][j] = a > 0.f ? a : 0.f;
    }
    __syncthreads();
    for (int i = tid; i < Bn * Cc; i += 256) {
        const int b = i >> 7, c = i & (Cc - 1);
        float z0 = 0.f, z1 = 0.f;
        #pragma unroll
        for (int j = 0; j < MID; ++j) {
            z0 += mid[b][j] * sk2[j * (2 * Cc) + c];
            z1 += mid[b][j] * sk2[j * (2 * Cc) + Cc + c];
        }
        const float m = fmaxf(z0, z1);
        const float e0 = __expf(z0 - m), e1 = __expf(z1 - m);
        const float inv = 1.f / (e0 + e1);
        wgt[b * (2 * Cc) + c]      = e0 * inv;
        wgt[b * (2 * Cc) + Cc + c] = e1 * inv;
    }
}

__global__ void k_combine(const float* __restrict__ fl, const float* __restrict__ fg,
                          const float* __restrict__ wgt, float* __restrict__ out) {
    int idx = blockIdx.x * blockDim.x + threadIdx.x;   // (b, c, hw), hw fastest
    int hw = idx & (HW - 1);
    int c  = (idx >> 12) & (Cc - 1);
    int b  = idx >> 19;
    const size_t ri = ((size_t)(b * HW + hw)) * Cc + c;
    out[idx] = wgt[b * (2 * Cc) + c] * fl[ri] + wgt[b * (2 * Cc) + Cc + c] * fg[ri];
}

// ------------------------------- launcher ----------------------------------

extern "C" void kernel_launch(void* const* d_in, const int* in_sizes, int n_in,
                              void* d_out, int out_size, void* d_ws, size_t ws_size,
                              hipStream_t stream) {
    // inputs in setup_inputs() dict order
    const float* x        = (const float*)d_in[0];
    const float* reduce_w = (const float*)d_in[1];
    const float* bn0_g = (const float*)d_in[2],  * bn0_b = (const float*)d_in[3];
    const float* bn0_m = (const float*)d_in[4],  * bn0_v = (const float*)d_in[5];
    const float* sp_g  = (const float*)d_in[6],  * sp_b  = (const float*)d_in[7];
    const float* sp_m  = (const float*)d_in[8],  * sp_v  = (const float*)d_in[9];
    const float* sh_g  = (const float*)d_in[10], * sh_b  = (const float*)d_in[11];
    const float* sh_m  = (const float*)d_in[12], * sh_v  = (const float*)d_in[13];
    const float* sv_g  = (const float*)d_in[14], * sv_b  = (const float*)d_in[15];
    const float* sv_m  = (const float*)d_in[16], * sv_v  = (const float*)d_in[17];
    const float* sp_w  = (const float*)d_in[18];
    const float* sh_w  = (const float*)d_in[19];
    const float* sv_w  = (const float*)d_in[20];
    const float* sf_w  = (const float*)d_in[21];
    const float* sf_b  = (const float*)d_in[22];
    const float* ln_g  = (const float*)d_in[23];
    const float* ln_b  = (const float*)d_in[24];
    const float* m_in_w    = (const float*)d_in[25];
    const float* m_conv_w  = (const float*)d_in[26];
    const float* m_conv_b  = (const float*)d_in[27];
    const float* m_xproj_w = (const float*)d_in[28];
    const float* m_dt_w    = (const float*)d_in[29];
    const float* m_dt_b    = (const float*)d_in[30];
    const float* m_Alog    = (const float*)d_in[31];
    const float* m_D       = (const float*)d_in[32];
    const float* m_out_w   = (const float*)d_in[33];
    const float* sk1_w     = (const float*)d_in[34];
    const float* sk2_w     = (const float*)d_in[35];
    float* out = (float*)d_out;

    // workspace bump allocator (deterministic layout every call)
    char* ws = (char*)d_ws;
    size_t off = 0;
    auto alloc = [&](size_t bytes) -> void* {
        void* p = ws + off;
        off = (off + bytes + 255) & ~(size_t)255;
        return p;
    };

    bf16_t* w_red_t = (bf16_t*)alloc((size_t)Cc * CIN * 2);       // [128][256]
    bf16_t* w_sp_t  = (bf16_t*)alloc((size_t)Cc * Cc * 2);        // [128][128]
    bf16_t* w_sf_t  = (bf16_t*)alloc((size_t)Cc * Cc * 2);        // [128][128]
    bf16_t* w_in_t  = (bf16_t*)alloc((size_t)(2 * DI) * Cc * 2);  // [512][128]
    bf16_t* w_xp_t  = (bf16_t*)alloc((size_t)(Rr + 2 * Nst) * DI * 2); // [40][256]
    bf16_t* w_out_t = (bf16_t*)alloc((size_t)Cc * DI * 2);        // [128][256]
    float* bn0_s = (float*)alloc(Cc * 4); float* bn0_o = (float*)alloc(Cc * 4);
    float* sp_s  = (float*)alloc(Cc * 4); float* sp_o  = (float*)alloc(Cc * 4);
    float* sh_s  = (float*)alloc(Cc * 4); float* sh_o  = (float*)alloc(Cc * 4);
    float* sv_s  = (float*)alloc(Cc * 4); float* sv_o  = (float*)alloc(Cc * 4);

    bf16_t* x_rows  = (bf16_t*)alloc((size_t)MTOK * CIN * 2);
    bf16_t* xr_rows = (bf16_t*)alloc((size_t)MTOK * Cc * 2);
    bf16_t* p_rows  = (bf16_t*)alloc((size_t)MTOK * Cc * 2);
    bf16_t* hv_rows = (bf16_t*)alloc((size_t)MTOK * Cc * 2);
    float*  fl      = (float*) alloc((size_t)MTOK * Cc * 4);
    bf16_t* xn      = (bf16_t*)alloc((size_t)MSEQ * Cc * 2);
    bf16_t* xz      = (bf16_t*)alloc((size_t)MSEQ * 2 * DI * 2);
    bf16_t* xc      = (bf16_t*)alloc((size_t)MSEQ * DI * 2);
    float*  dbl     = (float*) alloc((size_t)MSEQ * (Rr + 2 * Nst) * 4);
    bf16_t* yg      = (bf16_t*)alloc((size_t)MSEQ * DI * 2);
    float*  yout    = (float*) alloc((size_t)MSEQ * Cc * 4);
    float*  fg      = (float*) alloc((size_t)MTOK * Cc * 4);
    float*  s_vec   = (float*) alloc(Bn * Cc * 4);
    float*  wgt     = (float*) alloc(Bn * 2 * Cc * 4);
    (void)ws_size; (void)in_sizes; (void)n_in; (void)out_size;

    // ---- prep: BN fold + weights to bf16 (transposed where needed) ----
    k_bn_fold<<<1, 128, 0, stream>>>(bn0_g, bn0_b, bn0_m, bn0_v, bn0_s, bn0_o, Cc);
    k_bn_fold<<<1, 128, 0, stream>>>(sp_g, sp_b, sp_m, sp_v, sp_s, sp_o, Cc);
    k_bn_fold<<<1, 128, 0, stream>>>(sh_g, sh_b, sh_m, sh_v, sh_s, sh_o, Cc);
    k_bn_fold<<<1, 128, 0, stream>>>(sv_g, sv_b, sv_m, sv_v, sv_s, sv_o, Cc);

    k_cvt<<<(Cc * CIN + 255) / 256, 256, 0, stream>>>(reduce_w, w_red_t, Cc * CIN);
    k_cvt<<<(Cc * Cc + 255) / 256, 256, 0, stream>>>(sp_w, w_sp_t, Cc * Cc);
    k_cvt<<<(Cc * Cc + 255) / 256, 256, 0, stream>>>(sf_w, w_sf_t, Cc * Cc);
    k_cvt_T<<<(Cc * 2 * DI + 255) / 256, 256, 0, stream>>>(m_in_w, w_in_t, Cc, 2 * DI);
    k_cvt_T<<<(DI * 40 + 255) / 256, 256, 0, stream>>>(m_xproj_w, w_xp_t, DI, 40);
    k_cvt_T<<<(DI * Cc + 255) / 256, 256, 0, stream>>>(m_out_w, w_out_t, DI, Cc);

    // ---- local path ----
    k_nchw2rows<<<(MTOK * CIN) / 256, 256, 0, stream>>>(x, x_rows);
    gemm_bf16_k<EPI_BNRELU, true><<<dim3(MTOK / BM, Cc / BN), 256, 0, stream>>>(
        x_rows, w_red_t, xr_rows, bn0_s, bn0_o, MTOK, Cc, CIN);
    gemm_bf16_k<EPI_BNRELU, true><<<dim3(MTOK / BM, Cc / BN), 256, 0, stream>>>(
        xr_rows, w_sp_t, p_rows, sp_s, sp_o, MTOK, Cc, Cc);
    k_dwconv<<<(MTOK * Cc) / 256, 256, 0, stream>>>(
        p_rows, sh_w, sv_w, sh_s, sh_o, sv_s, sv_o, hv_rows);
    gemm_bf16_k<EPI_BIAS, false><<<dim3(MTOK / BM, Cc / BN), 256, 0, stream>>>(
        hv_rows, w_sf_t, fl, nullptr, sf_b, MTOK, Cc, Cc);

    // ---- global path: 4 directions batched ----
    k_gather_ln<<<MSEQ / 8, 256, 0, stream>>>(xr_rows, ln_g, ln_b, xn);
    gemm_bf16_k<EPI_NONE, true><<<dim3(MSEQ / BM, (2 * DI) / BN), 256, 0, stream>>>(
        xn, w_in_t, xz, nullptr, nullptr, MSEQ, 2 * DI, Cc);
    k_conv1d<<<(MSEQ * DI) / 256, 256, 0, stream>>>(xz, m_conv_w, m_conv_b, xc);
    gemm_bf16_k<EPI_NONE, false><<<dim3(MSEQ / BM, 1), 256, 0, stream>>>(
        xc, w_xp_t, dbl, nullptr, nullptr, MSEQ, Rr + 2 * Nst, DI);
    k_scan<<<4 * Bn, 256, 0, stream>>>(dbl, xc, xz, m_dt_w, m_dt_b, m_Alog, m_D, yg);
    gemm_bf16_k<EPI_NONE, false><<<dim3(MSEQ / BM, Cc / BN), 256, 0, stream>>>(
        yg, w_out_t, yout, nullptr, nullptr, MSEQ, Cc, DI);
    k_feat_global<<<(MTOK * Cc) / 256, 256, 0, stream>>>(yout, xr_rows, fg);

    // ---- SE head + combine ----
    k_smean<<<Bn * Cc, 256, 0, stream>>>(fl, fg, s_vec);
    k_se<<<1, 256, 0, stream>>>(s_vec, sk1_w, sk2_w, wgt);
    k_combine<<<(Bn * Cc * HW) / 256, 256, 0, stream>>>(fl, fg, wgt, out);
}
